// LearnedWLSSetTransformerCorrection_2671469658619
// MI455X (gfx1250) — matrix operand order repaired and checked by reading, not compiled, not run on hardware
//
#include <hip/hip_runtime.h>

// ---------------------------------------------------------------------------
// LearnedWLSSetTransformerCorrection for gfx1250 (MI455X, wave32, WMMA)
// One wave per batch. All GEMMs via v_wmma_f32_16x16x32_f16 with K padded to
// multiples of 32 so every operand load is an unconditional ds_load_b128.
// tile_mm2 computes both 16-row tiles per B-tile load (B reuse).
// B=32768, M=32, D=64, H=4, HD=16, FF=256
// ---------------------------------------------------------------------------

typedef __attribute__((ext_vector_type(16))) _Float16 v16h;
typedef __attribute__((ext_vector_type(8)))  _Float16 v8h;
typedef __attribute__((ext_vector_type(8)))  float    v8f;

#define BATCH   32768
#define SEQ     32
#define DIM     64
#define NHEAD   4
#define HDIM    16
#define FFDIM   256
#define WAVES   4
#define NBLK    2048
#define TOTW    (NBLK * WAVES)
#define ITERS   (BATCH / TOTW)          // 4
#define QKLD    128                     // head-padded row stride for q/k
#define POOLB   16384                   // per-wave overlay pool bytes

struct Params {
  const float* x;            // (B,32,4)
  const unsigned char* pad;  // (B,32) bool
  const float *embW, *embB;
  const float *Wq, *bq, *Wk, *bk, *Wv, *bv, *Wo, *bo;
  const float *ln1g, *ln1b;
  const float *W1, *b1, *W2, *b2;
  const float *ln2g, *ln2b;
  const float *Wout, *bout;
  const float *Wl, *bl, *al;
  const float *Wc, *bc, *ac;
  float* out;                // (B,4)
};

// ---- WMMA operand loaders (contiguous k order: k = k0 + 16*(lane>>4) + s, A and B alike) -------------------
__device__ __forceinline__ v16h ld_a16(const _Float16* s, int ld, int m0, int k0) {
  const int lane = threadIdx.x & 31;
  const int g = lane >> 4;
  const _Float16* row = s + (m0 + (lane & 15)) * ld;
  v8h t0 = *(const v8h*)(row + k0 + 16 * g);
  v8h t1 = *(const v8h*)(row + k0 + 16 * g + 8);
  v16h a;
#pragma unroll
  for (int i = 0; i < 8; ++i) { a[i] = t0[i]; a[8 + i] = t1[i]; }
  return a;
}

__device__ __forceinline__ v16h ld_bt16(const _Float16* s, int ldk, int n0, int k0) {
  const int lane = threadIdx.x & 31;
  const int g = lane >> 4;
  const _Float16* row = s + (n0 + (lane & 15)) * ldk + k0 + 16 * g;
  v8h t0 = *(const v8h*)(row);
  v8h t1 = *(const v8h*)(row + 8);
  v16h b;
#pragma unroll
  for (int i = 0; i < 8; ++i) { b[i] = t0[i]; b[8 + i] = t1[i]; }
  return b;
}

// C/D 16x16 f32: VGPR r holds row m0 + r + 8*(lane>>4), col n0 + (lane&15).
__device__ __forceinline__ void st_c16(_Float16* C, int ld, int m0, int n0, v8f c) {
  const int lane = threadIdx.x & 31;
  const int n = n0 + (lane & 15), g = lane >> 4;
#pragma unroll
  for (int r = 0; r < 8; ++r) C[(m0 + r + 8 * g) * ld + n] = (_Float16)c[r];
}
__device__ __forceinline__ void st_c32(float* C, int ld, int m0, int n0, v8f c) {
  const int lane = threadIdx.x & 31;
  const int n = n0 + (lane & 15), g = lane >> 4;
#pragma unroll
  for (int r = 0; r < 8; ++r) C[(m0 + r + 8 * g) * ld + n] = c[r];
}
__device__ __forceinline__ void addst_c32(float* C, int ld, int m0, int n0, v8f c) {
  const int lane = threadIdx.x & 31;
  const int n = n0 + (lane & 15), g = lane >> 4;
#pragma unroll
  for (int r = 0; r < 8; ++r) C[(m0 + r + 8 * g) * ld + n] += c[r];
}
__device__ __forceinline__ void st_cT16(_Float16* Ct, int ldT, int m0, int n0, v8f c) {
  const int lane = threadIdx.x & 31;
  const int n = n0 + (lane & 15), g = lane >> 4;
#pragma unroll
  for (int r = 0; r < 8; ++r) Ct[n * ldT + (m0 + r + 8 * g)] = (_Float16)c[r];
}

// Both 16-row output tiles for one column tile: load each B tile ONCE.
__device__ __forceinline__ void tile_mm2(const _Float16* A, int ldA,
                                         const _Float16* Bt, int ldB, int n0,
                                         int K, float init, v8f& c0, v8f& c1) {
#pragma unroll
  for (int r = 0; r < 8; ++r) { c0[r] = init; c1[r] = init; }
#pragma unroll
  for (int k0 = 0; k0 < K; k0 += 32) {
    v16h b  = ld_bt16(Bt, ldB, n0, k0);
    v16h a0 = ld_a16(A, ldA, 0,  k0);
    v16h a1 = ld_a16(A, ldA, 16, k0);
    c0 = __builtin_amdgcn_wmma_f32_16x16x32_f16(false, a0, false, b, (short)0, c0,
                                                false, false);
    c1 = __builtin_amdgcn_wmma_f32_16x16x32_f16(false, a1, false, b, (short)0, c1,
                                                false, false);
  }
}

// Register-resident layernorm of one 64-wide row (f32 in-place + f16 copy).
__device__ __forceinline__ void ln_row(float* hf, _Float16* h16,
                                       const float* g, const float* bt, int row) {
  float r[DIM];
  float4* base = (float4*)(hf + row * DIM);
#pragma unroll
  for (int q = 0; q < DIM / 4; ++q) {
    float4 t = base[q];
    r[4 * q] = t.x; r[4 * q + 1] = t.y; r[4 * q + 2] = t.z; r[4 * q + 3] = t.w;
  }
  float mu = 0.f;
#pragma unroll
  for (int k = 0; k < DIM; ++k) mu += r[k];
  mu *= (1.f / DIM);
  float var = 0.f;
#pragma unroll
  for (int k = 0; k < DIM; ++k) { float d = r[k] - mu; var += d * d; }
  var *= (1.f / DIM);
  const float rs = rsqrtf(var + 1e-5f);
#pragma unroll
  for (int k = 0; k < DIM; ++k) r[k] = (r[k] - mu) * rs * g[k] + bt[k];
#pragma unroll
  for (int q = 0; q < DIM / 4; ++q) {
    float4 t; t.x = r[4 * q]; t.y = r[4 * q + 1]; t.z = r[4 * q + 2]; t.w = r[4 * q + 3];
    base[q] = t;
  }
  v8h* d16 = (v8h*)(h16 + row * DIM);
#pragma unroll
  for (int q = 0; q < DIM / 8; ++q) {
    v8h t;
#pragma unroll
    for (int i = 0; i < 8; ++i) t[i] = (_Float16)r[8 * q + i];
    d16[q] = t;
  }
}

// Register-resident softmax over one 32-wide row of scores -> f16 attn row.
__device__ __forceinline__ void softmax_row(const float* ssc, _Float16* sat, int row) {
  float r[SEQ];
  const float4* src = (const float4*)(ssc + row * SEQ);
#pragma unroll
  for (int q = 0; q < SEQ / 4; ++q) {
    float4 t = src[q];
    r[4 * q] = t.x; r[4 * q + 1] = t.y; r[4 * q + 2] = t.z; r[4 * q + 3] = t.w;
  }
  float mx = r[0];
#pragma unroll
  for (int j = 1; j < SEQ; ++j) mx = fmaxf(mx, r[j]);
  float sum = 0.f;
#pragma unroll
  for (int j = 0; j < SEQ; ++j) { r[j] = __expf(r[j] - mx); sum += r[j]; }
  const float inv = 1.f / sum;
  v8h* dst = (v8h*)(sat + row * SEQ);
#pragma unroll
  for (int q = 0; q < SEQ / 8; ++q) {
    v8h t;
#pragma unroll
    for (int i = 0; i < 8; ++i) t[i] = (_Float16)(r[8 * q + i] * inv);
    dst[q] = t;
  }
}

__global__ __launch_bounds__(WAVES * 32, 1)
void wls_settrans_kernel(Params p) {
  // ---- shared weights, f16 transposed [N][K] ------------------------------
  __shared__ __align__(16) _Float16 w_embT[DIM * 32];
  __shared__ __align__(16) _Float16 w_qT[DIM * DIM];
  __shared__ __align__(16) _Float16 w_kT[DIM * DIM];
  __shared__ __align__(16) _Float16 w_vT[DIM * DIM];
  __shared__ __align__(16) _Float16 w_oT[DIM * DIM];
  __shared__ __align__(16) _Float16 w_1T[FFDIM * DIM];
  __shared__ __align__(16) _Float16 w_2T[DIM * FFDIM];
  // ---- per-wave persistent scratch ---------------------------------------
  __shared__ __align__(16) _Float16 s_x[WAVES][SEQ * 32];
  __shared__ __align__(16) _Float16 s_h16[WAVES][SEQ * DIM];
  __shared__ __align__(16) float    s_hf [WAVES][SEQ * DIM];
  __shared__ __align__(16) _Float16 s_qp[WAVES][SEQ * QKLD];
  __shared__ __align__(16) _Float16 s_kp[WAVES][SEQ * QKLD];
  __shared__ __align__(16) float    s_feat[WAVES][DIM];
  __shared__ __align__(16) float    s_fo[WAVES][DIM];
  // ---- per-wave overlay pool: attention scratch / FFN activations --------
  __shared__ __align__(16) unsigned char s_pool[WAVES][POOLB];

  const int tid = threadIdx.x;
  const int lane = tid & 31;
  const int wid = tid >> 5;

  // stage weights: global f32 (row-major [K][N]) -> LDS f16 transposed [N][K]
  for (int i = tid; i < DIM * 32; i += WAVES * 32) {
    int n = i >> 5, k = i & 31;
    w_embT[i] = (k < 4) ? (_Float16)p.embW[k * DIM + n] : (_Float16)0.f;
  }
  for (int i = tid; i < DIM * DIM; i += WAVES * 32) {
    int k = i >> 6, n = i & 63;
    w_qT[n * DIM + k] = (_Float16)p.Wq[i];
    w_kT[n * DIM + k] = (_Float16)p.Wk[i];
    w_vT[n * DIM + k] = (_Float16)p.Wv[i];
    w_oT[n * DIM + k] = (_Float16)p.Wo[i];
  }
  for (int i = tid; i < DIM * FFDIM; i += WAVES * 32) {
    int k1 = i >> 8, n1 = i & 255;          // W1: (64,256)
    w_1T[n1 * DIM + k1] = (_Float16)p.W1[i];
    int k2 = i >> 6, n2 = i & 63;           // W2: (256,64)
    w_2T[n2 * FFDIM + k2] = (_Float16)p.W2[i];
  }
  __syncthreads();

  _Float16* sx  = s_x[wid];   _Float16* sh16 = s_h16[wid]; float* shf = s_hf[wid];
  _Float16* sqp = s_qp[wid];  _Float16* skp  = s_kp[wid];
  float*    sfe = s_feat[wid]; float*   sfo  = s_fo[wid];
  unsigned char* pool = s_pool[wid];
  _Float16* svT = (_Float16*)(pool);              // 4 KB  (attention phase)
  _Float16* so  = (_Float16*)(pool + 4096);       // 4 KB
  float*    ssc = (float*)   (pool + 8192);       // 4 KB
  _Float16* sat = (_Float16*)(pool + 12288);      // 2 KB
  _Float16* sff = (_Float16*)(pool);              // 16 KB (FFN phase overlay)

  // one-time zero fill of persistent pad zones (wave-private)
  for (int i = lane; i < SEQ * 32; i += 32) sx[i] = (_Float16)0.f;
  for (int i = lane; i < SEQ * QKLD; i += 32) {
    if ((i & 31) >= HDIM) { sqp[i] = (_Float16)0.f; skp[i] = (_Float16)0.f; }
  }

  for (int t = 0; t < ITERS; ++t) {
    const int b = blockIdx.x * WAVES + wid + t * TOTW;
    if (t + 1 < ITERS)  // gfx1250 global_prefetch_b8 for next batch's x
      __builtin_prefetch(p.x + (size_t)(b + TOTW) * SEQ * 4, 0, 1);

    const unsigned char* pmb = p.pad + (size_t)b * SEQ;
    const float4 x4 = *(const float4*)(p.x + (size_t)b * SEQ * 4 + lane * 4);
    const float xr[4] = { x4.x, x4.y, x4.z, x4.w };
    {
      v8h xs;
      xs[0] = (_Float16)xr[0]; xs[1] = (_Float16)xr[1];
      xs[2] = (_Float16)xr[2]; xs[3] = (_Float16)xr[3];
      xs[4] = (_Float16)0.f; xs[5] = (_Float16)0.f;
      xs[6] = (_Float16)0.f; xs[7] = (_Float16)0.f;
      *(v8h*)(sx + lane * 32) = xs;    // cols 8..31 stay zero from init
    }
    const int pmi = pmb[lane] ? 1 : 0;
    const unsigned validmask = (unsigned)__ballot(!pmi);
    // per-lane score-column masks (column = nt*16 + (lane&15))
    const bool mcol0 = pmb[lane & 15] != 0;
    const bool mcol1 = pmb[16 + (lane & 15)] != 0;

    v8f c0, c1;
    // ---- h = x @ embW + emb_b  (K padded 4 -> 32) -------------------------
#pragma unroll
    for (int nt = 0; nt < 4; ++nt) {
      const int n0 = nt * 16;
      tile_mm2(sx, 32, w_embT, 32, n0, 32, p.embB[n0 + (lane & 15)], c0, c1);
      st_c16(sh16, DIM, 0, n0, c0);  st_c32(shf, DIM, 0, n0, c0);
      st_c16(sh16, DIM, 16, n0, c1); st_c32(shf, DIM, 16, n0, c1);
    }

    // ---- q, k (head-padded row-major), v (transposed) ---------------------
#pragma unroll
    for (int nt = 0; nt < 4; ++nt) {
      const int n0 = nt * 16, nn = n0 + (lane & 15);
      tile_mm2(sh16, DIM, w_qT, DIM, n0, DIM, p.bq[nn], c0, c1);
      st_c16(sqp, QKLD, 0, nt * 32, c0);
      st_c16(sqp, QKLD, 16, nt * 32, c1);
      tile_mm2(sh16, DIM, w_kT, DIM, n0, DIM, p.bk[nn], c0, c1);
      st_c16(skp, QKLD, 0, nt * 32, c0);
      st_c16(skp, QKLD, 16, nt * 32, c1);
      tile_mm2(sh16, DIM, w_vT, DIM, n0, DIM, p.bv[nn], c0, c1);
      st_cT16(svT, SEQ, 0, n0, c0);
      st_cT16(svT, SEQ, 16, n0, c1);
    }

    // ---- attention per head (score K padded 16 -> 32 via head padding) ----
#pragma unroll
    for (int hh = 0; hh < NHEAD; ++hh) {
#pragma unroll
      for (int nt = 0; nt < 2; ++nt) {
        tile_mm2(sqp + hh * 32, QKLD, skp + hh * 32, QKLD, nt * 16, 32, 0.f, c0, c1);
        const int n = nt * 16 + (lane & 15), g = lane >> 4;
        const bool masked = nt == 0 ? mcol0 : mcol1;
#pragma unroll
        for (int r = 0; r < 8; ++r) {
          ssc[(r + 8 * g) * SEQ + n]      = masked ? -1.0e9f : c0[r] * 0.25f;
          ssc[(16 + r + 8 * g) * SEQ + n] = masked ? -1.0e9f : c1[r] * 0.25f;
        }
      }
      softmax_row(ssc, sat, lane);
      // o_h = attn @ v_h   (K = 32 exact)
      tile_mm2(sat, SEQ, svT + hh * HDIM * SEQ, SEQ, 0, SEQ, 0.f, c0, c1);
      st_c16(so, DIM, 0, hh * HDIM, c0);
      st_c16(so, DIM, 16, hh * HDIM, c1);
    }

    // ---- o @ Wo + bo, residual into shf, LN1 ------------------------------
#pragma unroll
    for (int nt = 0; nt < 4; ++nt) {
      const int n0 = nt * 16;
      tile_mm2(so, DIM, w_oT, DIM, n0, DIM, p.bo[n0 + (lane & 15)], c0, c1);
      addst_c32(shf, DIM, 0, n0, c0);
      addst_c32(shf, DIM, 16, n0, c1);
    }
    ln_row(shf, sh16, p.ln1g, p.ln1b, lane);

    // ---- FFN: relu(h@W1+b1)@W2+b2, residual, LN2 --------------------------
#pragma unroll
    for (int nt = 0; nt < FFDIM / 16; ++nt) {
      const int n0 = nt * 16;
      tile_mm2(sh16, DIM, w_1T, DIM, n0, DIM, p.b1[n0 + (lane & 15)], c0, c1);
#pragma unroll
      for (int r = 0; r < 8; ++r) {
        c0[r] = fmaxf(c0[r], 0.f);
        c1[r] = fmaxf(c1[r], 0.f);
      }
      st_c16(sff, FFDIM, 0, n0, c0);
      st_c16(sff, FFDIM, 16, n0, c1);
    }
#pragma unroll
    for (int nt = 0; nt < 4; ++nt) {
      const int n0 = nt * 16;
      tile_mm2(sff, FFDIM, w_2T, FFDIM, n0, FFDIM, p.b2[n0 + (lane & 15)], c0, c1);
      addst_c32(shf, DIM, 0, n0, c0);
      addst_c32(shf, DIM, 16, n0, c1);
    }
    ln_row(shf, sh16, p.ln2g, p.ln2b, lane);

    // ---- masked mean over rows -> feat; feat @ Wout + bout ----------------
    const int cnt = __popc(validmask);
    const float denom = fmaxf((float)cnt, 1.f);
#pragma unroll
    for (int cc = 0; cc < 2; ++cc) {
      const int col = lane + 32 * cc;
      float s = 0.f;
#pragma unroll
      for (int m = 0; m < SEQ; ++m) {
        const float bitf = (float)((validmask >> m) & 1u);
        s += bitf * shf[m * DIM + col];
      }
      sfe[col] = s / denom;
    }
#pragma unroll
    for (int cc = 0; cc < 2; ++cc) {
      const int col = lane + 32 * cc;
      float a = p.bout[col];
      for (int d = 0; d < DIM; ++d) a += sfe[d] * p.Wout[d * DIM + col];
      sfo[col] = a;
    }

    // ---- per-row WLS weight, lane = row -----------------------------------
    float acc = p.bl[0];
#pragma unroll
    for (int c = 0; c < 4; ++c) acc += xr[c] * p.Wl[c];
    for (int d = 0; d < DIM; ++d) acc += sfo[d] * p.Wl[4 + d];
    const float alc = p.al[0];
    float wv = acc >= 0.f ? acc : alc * acc;
    if (pmi) wv = 0.f;
    const float w2 = wv * wv;
    const float Ai[4] = { -xr[1], -xr[2], -xr[3], 1.f };
    const float r0 = xr[0];
    float Nm[4][4], yv[4];
#pragma unroll
    for (int i = 0; i < 4; ++i) {
      yv[i] = w2 * Ai[i] * r0;
#pragma unroll
      for (int j = 0; j < 4; ++j) Nm[i][j] = w2 * Ai[i] * Ai[j];
    }
    // butterfly reduce across 32 lanes
#pragma unroll
    for (int i = 0; i < 4; ++i) {
#pragma unroll
      for (int off = 16; off; off >>= 1) yv[i] += __shfl_xor(yv[i], off);
#pragma unroll
      for (int j = 0; j < 4; ++j)
#pragma unroll
        for (int off = 16; off; off >>= 1) Nm[i][j] += __shfl_xor(Nm[i][j], off);
    }
#pragma unroll
    for (int i = 0; i < 4; ++i) Nm[i][i] += 1e-12f;
    // SPD 4x4 Gaussian elimination (no pivot), fully unrolled
#pragma unroll
    for (int kk = 0; kk < 4; ++kk) {
      const float inv = 1.f / Nm[kk][kk];
#pragma unroll
      for (int i = kk + 1; i < 4; ++i) {
        const float f = Nm[i][kk] * inv;
#pragma unroll
        for (int j = kk; j < 4; ++j) Nm[i][j] -= f * Nm[kk][j];
        yv[i] -= f * yv[kk];
      }
    }
    float sol[4];
#pragma unroll
    for (int kk = 3; kk >= 0; --kk) {
      float s = yv[kk];
#pragma unroll
      for (int j = kk + 1; j < 4; ++j) s -= Nm[kk][j] * sol[j];
      sol[kk] = s / Nm[kk][kk];
    }

    // ---- corrections and output ------------------------------------------
    if (lane < 4) {
      const float acr = p.ac[0];
      float c2 = p.bc[lane];
      for (int d = 0; d < DIM; ++d) {
        float f = sfo[d];
        f = f >= 0.f ? f : acr * f;
        c2 += f * p.Wc[d * 4 + lane];
      }
      const float sv = lane == 0 ? sol[0] : lane == 1 ? sol[1]
                     : lane == 2 ? sol[2] : sol[3];
      p.out[(size_t)b * 4 + lane] = sv + c2;
    }
  }
}

extern "C" void kernel_launch(void* const* d_in, const int* in_sizes, int n_in,
                              void* d_out, int out_size, void* d_ws, size_t ws_size,
                              hipStream_t stream) {
  (void)in_sizes; (void)n_in; (void)d_ws; (void)ws_size; (void)out_size;
  Params p;
  p.x    = (const float*)d_in[0];
  p.pad  = (const unsigned char*)d_in[1];
  p.embW = (const float*)d_in[2];   p.embB = (const float*)d_in[3];
  p.Wq   = (const float*)d_in[4];   p.bq   = (const float*)d_in[5];
  p.Wk   = (const float*)d_in[6];   p.bk   = (const float*)d_in[7];
  p.Wv   = (const float*)d_in[8];   p.bv   = (const float*)d_in[9];
  p.Wo   = (const float*)d_in[10];  p.bo   = (const float*)d_in[11];
  p.ln1g = (const float*)d_in[12];  p.ln1b = (const float*)d_in[13];
  p.W1   = (const float*)d_in[14];  p.b1   = (const float*)d_in[15];
  p.W2   = (const float*)d_in[16];  p.b2   = (const float*)d_in[17];
  p.ln2g = (const float*)d_in[18];  p.ln2b = (const float*)d_in[19];
  p.Wout = (const float*)d_in[20];  p.bout = (const float*)d_in[21];
  p.Wl   = (const float*)d_in[22];  p.bl   = (const float*)d_in[23];
  p.al   = (const float*)d_in[24];
  p.Wc   = (const float*)d_in[25];  p.bc   = (const float*)d_in[26];
  p.ac   = (const float*)d_in[27];
  p.out  = (float*)d_out;
  wls_settrans_kernel<<<NBLK, WAVES * 32, 0, stream>>>(p);
}
